// LlamaAttention_19971597927167
// MI455X (gfx1250) — matrix operand order repaired and checked by reading, not compiled, not run on hardware
//
#include <hip/hip_runtime.h>

typedef __bf16 bf16;
typedef __attribute__((ext_vector_type(16))) bf16  v16bf;
typedef __attribute__((ext_vector_type(8)))  bf16  v8bf;
typedef __attribute__((ext_vector_type(8)))  float v8f;
typedef __attribute__((ext_vector_type(4)))  int   v4i;

#define BDIM 2
#define SDIM 1024
#define HID  4096
#define NHEAD 32
#define HDIM 128
#define MTOT (BDIM * SDIM)   // 2048

#if __has_builtin(__builtin_amdgcn_global_load_async_to_lds_b128)
#define HAVE_ASYNC_LDS 1
#else
#define HAVE_ASYNC_LDS 0
#endif

typedef v4i __attribute__((address_space(1)))* gv4i_p;   // global v4i*
typedef v4i __attribute__((address_space(3)))* lv4i_p;   // LDS v4i*

// ---------------------------------------------------------------------------
// fp32 -> bf16 conversion
// ---------------------------------------------------------------------------
__global__ void cvt_f32_bf16(const float* __restrict__ in, bf16* __restrict__ out, int n) {
    int t = blockIdx.x * blockDim.x + threadIdx.x;
    if (t < n) out[t] = (bf16)in[t];
}

// ---------------------------------------------------------------------------
// GEMM: C[m,n] = sum_k A[m,k] * W[n,k] + bias[n]
// A: [Mdim,Kdim] bf16, W: [Ndim,Kdim] bf16, C fp32.
// Block = 256 threads = 8 waves; wave computes 32x32 via 2x2 WMMA tiles.
// Block tile: 128 (M) x 64 (N).
// ---------------------------------------------------------------------------
__global__ __launch_bounds__(256)
void gemm_bf16_wmma(const bf16* __restrict__ A, const bf16* __restrict__ W,
                    const float* __restrict__ bias, float* __restrict__ C,
                    int Mdim, int Ndim, int Kdim) {
    const int lane = threadIdx.x & 31;
    const int wave = threadIdx.x >> 5;
    const int wm = wave & 3, wn = wave >> 2;
    const int m0 = blockIdx.y * 128 + wm * 32;
    const int n0 = blockIdx.x * 64  + wn * 32;
    const int r  = lane & 15, hi = lane >> 4;

    v8f acc[2][2] = {};

    for (int k0 = 0; k0 < Kdim; k0 += 32) {
        v16bf a[2], b[2];
#pragma unroll
        for (int i = 0; i < 2; ++i) {
            // A fragment (16x32): lane=row, k = hi*16 + [0..15] contiguous
            const bf16* pa = A + (size_t)(m0 + 16 * i + r) * Kdim + k0 + hi * 16;
            v8bf lo = *(const v8bf*)pa;
            v8bf up = *(const v8bf*)(pa + 8);
#pragma unroll
            for (int e = 0; e < 8; ++e) { a[i][e] = lo[e]; a[i][8 + e] = up[e]; }
            // B fragment (32x16): lane=col(n), k = hi*16 + [0..15] contiguous
            const bf16* pb = W + (size_t)(n0 + 16 * i + r) * Kdim + k0 + hi * 16;
            b[i] = *(const v16bf*)pb;
            if (k0 + 128 < Kdim) {                 // global_prefetch_b8 next tiles
                __builtin_prefetch(pa + 128, 0, 0);
                __builtin_prefetch(pb + 128, 0, 0);
            }
        }
#pragma unroll
        for (int i = 0; i < 2; ++i)
#pragma unroll
            for (int j = 0; j < 2; ++j)
                acc[i][j] = __builtin_amdgcn_wmma_f32_16x16x32_bf16(
                    false, a[i], false, b[j], (short)0, acc[i][j], false, false);
    }

#pragma unroll
    for (int i = 0; i < 2; ++i)
#pragma unroll
        for (int j = 0; j < 2; ++j) {
            const int col = n0 + 16 * j + r;
            const float bv = bias ? bias[col] : 0.0f;
#pragma unroll
            for (int e = 0; e < 8; ++e) {
                const int row = m0 + 16 * i + 8 * hi + e;
                C[(size_t)row * Ndim + col] = acc[i][j][e] + bv;
            }
        }
}

// ---------------------------------------------------------------------------
// RoPE (interleaved) + repack [B,S,NH,HD] fp32 -> [B,NH,S,HD] bf16 for Q and K
// ---------------------------------------------------------------------------
__global__ void rope_repack(const float* __restrict__ Qf, const float* __restrict__ Kf,
                            const int* __restrict__ pos_ids,
                            bf16* __restrict__ Qb, bf16* __restrict__ Kb) {
    const int t = blockIdx.x * blockDim.x + threadIdx.x;
    if (t >= BDIM * SDIM * NHEAD * (HDIM / 2)) return;
    const int d2 = t & 63;
    const int h  = (t >> 6) & 31;
    const int s  = (t >> 11) & 1023;
    const int b  = t >> 21;

    const float pos = (float)pos_ids[b * SDIM + s];
    // inv_freq = 10000^{-2*d2/128} = exp(-d2 * ln(10000)/64)
    const float ang = pos * __expf(-0.14390972160f * (float)d2);
    const float c = __cosf(ang), sn = __sinf(ang);

    const size_t iin  = (size_t)(b * SDIM + s) * HID + h * HDIM + 2 * d2;
    const size_t iout = ((size_t)(b * NHEAD + h) * SDIM + s) * HDIM + 2 * d2;

    const float q0 = Qf[iin], q1 = Qf[iin + 1];
    Qb[iout]     = (bf16)(q0 * c - q1 * sn);
    Qb[iout + 1] = (bf16)(q1 * c + q0 * sn);
    const float k0 = Kf[iin], k1 = Kf[iin + 1];
    Kb[iout]     = (bf16)(k0 * c - k1 * sn);
    Kb[iout + 1] = (bf16)(k1 * c + k0 * sn);
}

// V repack [B,S,NH,HD] fp32 -> [B,NH,S,HD] bf16
__global__ void v_repack(const float* __restrict__ Vf, bf16* __restrict__ Vb) {
    const int t = blockIdx.x * blockDim.x + threadIdx.x;
    if (t >= MTOT * HID) return;
    const int d = t & 127;
    const int h = (t >> 7) & 31;
    const int s = (t >> 12) & 1023;
    const int b = t >> 22;
    Vb[((size_t)(b * NHEAD + h) * SDIM + s) * HDIM + d] =
        (bf16)Vf[(size_t)(b * SDIM + s) * HID + h * HDIM + d];
}

// ---------------------------------------------------------------------------
// Flash attention (causal). Grid: (S/128, NH, B), block 256 = 8 waves.
// Each wave owns 16 query rows; K/V tiles of 32 keys staged in LDS
// (K via async global->LDS copy when available; V transposed in-register).
// Output written bf16 directly back in [B,S,NH,HD] (= [M,HID]) layout.
// ---------------------------------------------------------------------------
__global__ __launch_bounds__(256)
void flash_attn(const bf16* __restrict__ Qb, const bf16* __restrict__ Kb,
                const bf16* __restrict__ Vb, bf16* __restrict__ Ob) {
    __shared__ bf16 sK[32 * 128];       // [key][d]
    __shared__ bf16 sVt[128 * 32];      // [d][key]   (transposed for PV B-frags)
    __shared__ bf16 sP[8][16 * 32];     // per-wave P scratch [row][key]

    const int lane = threadIdx.x & 31, wave = threadIdx.x >> 5;
    const int r = lane & 15, hi = lane >> 4;
    const int b = blockIdx.z, h = blockIdx.y;
    const size_t base = (size_t)(b * NHEAD + h) * SDIM * HDIM;
    const int qblk  = blockIdx.x * 128;
    const int qrow0 = qblk + wave * 16;

    // Q fragments: 16 rows x 128 d  -> 4 A-frags of 16x32
    v16bf aq[4];
#pragma unroll
    for (int kk = 0; kk < 4; ++kk) {
        const bf16* p = Qb + base + (size_t)(qrow0 + r) * HDIM + kk * 32 + hi * 16;
        v8bf lo = *(const v8bf*)p;
        v8bf up = *(const v8bf*)(p + 8);
#pragma unroll
        for (int e = 0; e < 8; ++e) { aq[kk][e] = lo[e]; aq[kk][8 + e] = up[e]; }
    }

    v8f o[8] = {};
    float mrun[8], lrun[8];
#pragma unroll
    for (int e = 0; e < 8; ++e) { mrun[e] = -1e30f; lrun[e] = 0.0f; }

    const int nkt = blockIdx.x * 4 + 4;       // causal: keys <= qblk+127
    const int ldr = threadIdx.x >> 3;         // 0..31  key row for coop load
    const int lds = (threadIdx.x & 7) * 16;   // d offset for coop load

    for (int kt = 0; kt < nkt; ++kt) {
        const int k0 = kt * 32;
        const bf16* gk = Kb + base + (size_t)(k0 + ldr) * HDIM + lds;
        const bf16* gv = Vb + base + (size_t)(k0 + ldr) * HDIM + lds;
        if (kt + 1 < nkt) {                           // prefetch next tile
            __builtin_prefetch(gk + 32 * HDIM, 0, 0);
            __builtin_prefetch(gv + 32 * HDIM, 0, 0);
        }
#if HAVE_ASYNC_LDS
        // K tile: pure copy, consumed row-major -> async global->LDS (ASYNCcnt)
        {
            gv4i_p g1 = (gv4i_p)(bf16*)gk;
            lv4i_p l1 = (lv4i_p)&sK[ldr * 128 + lds];
            __builtin_amdgcn_global_load_async_to_lds_b128(g1, l1, 0, 0);
            __builtin_amdgcn_global_load_async_to_lds_b128(g1, l1, 16, 0);
        }
#else
        *(v16bf*)&sK[ldr * 128 + lds] = *(const v16bf*)gk;
#endif
        // V tile: needs transpose -> register round-trip
        v16bf vv = *(const v16bf*)gv;
#pragma unroll
        for (int e = 0; e < 16; ++e) sVt[(lds + e) * 32 + ldr] = vv[e];
#if HAVE_ASYNC_LDS
#if __has_builtin(__builtin_amdgcn_s_wait_asynccnt)
        __builtin_amdgcn_s_wait_asynccnt(0);
#else
        asm volatile("s_wait_asynccnt 0" ::: "memory");
#endif
#endif
        __syncthreads();

        if (k0 <= qrow0 + 15) {   // wave-uniform: EXEC stays all-ones for WMMA
            // scores: 16 q-rows x 32 keys, contraction over d=128
            v8f c0 = {}, c1 = {};
#pragma unroll
            for (int kk = 0; kk < 4; ++kk) {
                v16bf bk0 = *(const v16bf*)&sK[r * 128 + kk * 32 + hi * 16];
                v16bf bk1 = *(const v16bf*)&sK[(16 + r) * 128 + kk * 32 + hi * 16];
                c0 = __builtin_amdgcn_wmma_f32_16x16x32_bf16(false, aq[kk], false, bk0,
                                                             (short)0, c0, false, false);
                c1 = __builtin_amdgcn_wmma_f32_16x16x32_bf16(false, aq[kk], false, bk1,
                                                             (short)0, c1, false, false);
            }
            // online softmax over the 32 keys
            const float scale = 0.08838834764831845f;   // 1/sqrt(128)
            float p0[8], p1[8], fac[8];
#pragma unroll
            for (int e = 0; e < 8; ++e) {
                const int row  = qrow0 + 8 * hi + e;
                const int key0 = k0 + r;
                float s0 = c0[e] * scale; if (key0 > row)      s0 = -1e30f;
                float s1 = c1[e] * scale; if (key0 + 16 > row) s1 = -1e30f;
                float mx = fmaxf(s0, s1);
                mx = fmaxf(mx, __shfl_xor(mx, 1, 32));
                mx = fmaxf(mx, __shfl_xor(mx, 2, 32));
                mx = fmaxf(mx, __shfl_xor(mx, 4, 32));
                mx = fmaxf(mx, __shfl_xor(mx, 8, 32));
                const float mnew = fmaxf(mrun[e], mx);
                p0[e] = __expf(s0 - mnew);
                p1[e] = __expf(s1 - mnew);
                float rs = p0[e] + p1[e];
                rs += __shfl_xor(rs, 1, 32);
                rs += __shfl_xor(rs, 2, 32);
                rs += __shfl_xor(rs, 4, 32);
                rs += __shfl_xor(rs, 8, 32);
                fac[e]  = __expf(mrun[e] - mnew);
                lrun[e] = lrun[e] * fac[e] + rs;
                mrun[e] = mnew;
            }
#pragma unroll
            for (int f = 0; f < 8; ++f)
#pragma unroll
                for (int e = 0; e < 8; ++e) o[f][e] *= fac[e];

            // re-fragment P (C-layout -> A-layout) through per-wave LDS
            bf16* pw = sP[wave];
#pragma unroll
            for (int e = 0; e < 8; ++e) {
                const int rr = 8 * hi + e;
                pw[rr * 32 + r]      = (bf16)p0[e];
                pw[rr * 32 + 16 + r] = (bf16)p1[e];
            }
            __builtin_amdgcn_wave_barrier();   // same-wave LDS ordering fence
            v16bf pa;
            {
                const bf16* pp = pw + r * 32 + hi * 16;
                v8bf lo = *(const v8bf*)pp;
                v8bf up = *(const v8bf*)(pp + 8);
#pragma unroll
                for (int e = 0; e < 8; ++e) { pa[e] = lo[e]; pa[8 + e] = up[e]; }
            }
            // O += P @ V  (contraction over 32 keys, 8 d-column frags)
#pragma unroll
            for (int f = 0; f < 8; ++f) {
                v16bf bv = *(const v16bf*)&sVt[(f * 16 + r) * 32 + hi * 16];
                o[f] = __builtin_amdgcn_wmma_f32_16x16x32_bf16(false, pa, false, bv,
                                                               (short)0, o[f], false, false);
            }
        }
        __syncthreads();
    }

    // normalize + store in (b,s,h,d) layout for the output projection GEMM
#pragma unroll
    for (int e = 0; e < 8; ++e) lrun[e] = 1.0f / lrun[e];
#pragma unroll
    for (int f = 0; f < 8; ++f) {
        const int d = f * 16 + r;
#pragma unroll
        for (int e = 0; e < 8; ++e) {
            const int srow = qrow0 + 8 * hi + e;
            Ob[(size_t)(b * SDIM + srow) * HID + h * HDIM + d] = (bf16)(o[f][e] * lrun[e]);
        }
    }
}

// ---------------------------------------------------------------------------
extern "C" void kernel_launch(void* const* d_in, const int* in_sizes, int n_in,
                              void* d_out, int out_size, void* d_ws, size_t ws_size,
                              hipStream_t stream) {
    (void)in_sizes; (void)n_in; (void)out_size; (void)ws_size;
    const float* x  = (const float*)d_in[0];
    const float* Wq = (const float*)d_in[1];
    const float* bq = (const float*)d_in[2];
    const float* Wk = (const float*)d_in[3];
    const float* bk = (const float*)d_in[4];
    const float* Wv = (const float*)d_in[5];
    const float* bv = (const float*)d_in[6];
    const float* Wo = (const float*)d_in[7];
    const float* bo = (const float*)d_in[8];
    const int*   pos = (const int*)d_in[9];
    float* out = (float*)d_out;

    const size_t nX = (size_t)MTOT * HID;    // 8M
    const size_t nW = (size_t)HID * HID;     // 16M

    char* p = (char*)d_ws;
    bf16*  xb  = (bf16*)p;  p += nX * sizeof(bf16);
    bf16*  Wqb = (bf16*)p;  p += nW * sizeof(bf16);
    bf16*  Wkb = (bf16*)p;  p += nW * sizeof(bf16);
    bf16*  Wvb = (bf16*)p;  p += nW * sizeof(bf16);
    bf16*  Wob = (bf16*)p;  p += nW * sizeof(bf16);
    float* Qf  = (float*)p; p += nX * sizeof(float);
    float* Kf  = (float*)p; p += nX * sizeof(float);
    float* Vf  = (float*)p; p += nX * sizeof(float);
    bf16*  Qbh = (bf16*)p;  p += nX * sizeof(bf16);
    bf16*  Kbh = (bf16*)p;  p += nX * sizeof(bf16);
    bf16*  Vbh = (bf16*)p;  p += nX * sizeof(bf16);
    bf16*  Ob  = (bf16*)p;  p += nX * sizeof(bf16);

    // 1) downcast inputs to bf16
    cvt_f32_bf16<<<(nX + 255) / 256, 256, 0, stream>>>(x,  xb,  (int)nX);
    cvt_f32_bf16<<<(nW + 255) / 256, 256, 0, stream>>>(Wq, Wqb, (int)nW);
    cvt_f32_bf16<<<(nW + 255) / 256, 256, 0, stream>>>(Wk, Wkb, (int)nW);
    cvt_f32_bf16<<<(nW + 255) / 256, 256, 0, stream>>>(Wv, Wvb, (int)nW);
    cvt_f32_bf16<<<(nW + 255) / 256, 256, 0, stream>>>(Wo, Wob, (int)nW);

    // 2) Q/K/V projections (WMMA GEMMs)
    dim3 gg(HID / 64, MTOT / 128);
    gemm_bf16_wmma<<<gg, 256, 0, stream>>>(xb, Wqb, bq, Qf, MTOT, HID, HID);
    gemm_bf16_wmma<<<gg, 256, 0, stream>>>(xb, Wkb, bk, Kf, MTOT, HID, HID);
    gemm_bf16_wmma<<<gg, 256, 0, stream>>>(xb, Wvb, bv, Vf, MTOT, HID, HID);

    // 3) RoPE + head-major repack
    const int nrope = BDIM * SDIM * NHEAD * (HDIM / 2);
    rope_repack<<<(nrope + 255) / 256, 256, 0, stream>>>(Qf, Kf, pos, Qbh, Kbh);
    v_repack<<<(nX + 255) / 256, 256, 0, stream>>>(Vf, Vbh);

    // 4) causal flash attention (WMMA + async LDS staging)
    dim3 ga(SDIM / 128, NHEAD, BDIM);
    flash_attn<<<ga, 256, 0, stream>>>(Qbh, Kbh, Vbh, Ob);

    // 5) output projection
    gemm_bf16_wmma<<<gg, 256, 0, stream>>>(Ob, Wob, bo, out, MTOT, HID, HID);
}
